// RSEBERTAttention_2113123910069
// MI455X (gfx1250) — matrix operand order repaired and checked by reading, not compiled, not run on hardware
//
#include <hip/hip_runtime.h>
#include <hip/hip_bf16.h>

typedef __attribute__((ext_vector_type(16))) _Float16 v16h;
typedef __attribute__((ext_vector_type(8)))  float    v8f;

#define HID   768
#define NHEAD 12
#define HD    64
#define SEQ   1024
#define BATCH 16
#define MTOT  (BATCH * SEQ)   // 16384

union Frag16 { v16h h; uint4 u[2]; };

__device__ __forceinline__ v8f wmma_f16(const Frag16& a, const Frag16& b, v8f c) {
  // D = A(16x32 f16) * B(32x16 f16) + C(16x16 f32)
  return __builtin_amdgcn_wmma_f32_16x16x32_f16(false, a.h, false, b.h,
                                                (short)0, c, false, false);
}

// -------- async global->LDS copy (16B per lane), with synchronous fallback
#if __has_builtin(__builtin_amdgcn_global_load_async_to_lds_b128) && \
    __has_builtin(__builtin_amdgcn_s_wait_asynccnt)
#define HAVE_ASYNC_LDS 1
#else
#define HAVE_ASYNC_LDS 0
#endif

typedef int v4i_vs __attribute__((vector_size(16)));
typedef __attribute__((address_space(1))) v4i_vs* gas_p;   // global
typedef __attribute__((address_space(3))) v4i_vs* las_p;   // LDS

__device__ __forceinline__ void async_copy16(void* lds, const void* gsrc) {
#if HAVE_ASYNC_LDS
  __builtin_amdgcn_global_load_async_to_lds_b128(
      (gas_p)(const_cast<void*>(gsrc)), (las_p)(lds), 0, 0);
#else
  *(uint4*)lds = *(const uint4*)gsrc;
#endif
}

#if HAVE_ASYNC_LDS
#define WAIT_ASYNC(n) __builtin_amdgcn_s_wait_asynccnt(n)
#else
#define WAIT_ASYNC(n)
#endif

// ---------------------------------------------------------------- f32 -> f16
__global__ __launch_bounds__(256) void cvt_f32_f16(const float* __restrict__ in,
                                                   _Float16* __restrict__ out,
                                                   int n4) {
  int i = blockIdx.x * blockDim.x + threadIdx.x;
  if (i >= n4) return;
  float4 v = ((const float4*)in)[i];
  union { _Float16 h[4]; uint2 u; } o;
  o.h[0] = (_Float16)v.x; o.h[1] = (_Float16)v.y;
  o.h[2] = (_Float16)v.z; o.h[3] = (_Float16)v.w;
  ((uint2*)out)[i] = o.u;
}

// --------------------------------------------- fused QKV GEMM + bias + RoPE
// Block: 8 waves share one N-group (= one head, 64 cols) across 128 rows.
// Weight K-slices (3 x 64 x 32 f16 = 12KB) are double-buffered in LDS via
// async DMA. Outputs: qr,kr f16 [B,H,S,D] (RoPE'd, q scaled); vt f16 [B,H,D,S].
__global__ __launch_bounds__(256) void qkv_rope_kernel(
    const _Float16* __restrict__ xh,
    const _Float16* __restrict__ wq, const _Float16* __restrict__ wk,
    const _Float16* __restrict__ wv,
    const float* __restrict__ bq, const float* __restrict__ bk,
    const float* __restrict__ bv,
    const float* __restrict__ cosc, const float* __restrict__ sinc,
    _Float16* __restrict__ qr, _Float16* __restrict__ kr,
    _Float16* __restrict__ vt) {
  __shared__ __align__(16) _Float16 wl[2][3][64][32];   // 24 KB

  const int tid  = threadIdx.x;
  const int lane = tid & 31;
  const int wid  = tid >> 5;
  const int ng   = blockIdx.x % (HID / 64);     // head / N-group
  const int m0   = (blockIdx.x / (HID / 64) * 8 + wid) * 16;
  const int n0   = ng * 64;
  const int l15  = lane & 15;
  const int hi   = lane >> 4;

  const _Float16* const wmat[3] = {wq, wk, wv};

  // fill one K-slice: 768 16B-chunks, 3 per thread
  auto fill = [&](int buf, int kb) {
#pragma unroll
    for (int i = 0; i < 3; ++i) {
      const int c  = tid + 256 * i;
      const int p  = c >> 8;
      const int r  = c & 255;
      const int nl = r >> 2;
      const int ch = r & 3;
      async_copy16(&wl[buf][p][nl][ch * 8],
                   wmat[p] + (size_t)(n0 + nl) * HID + kb + ch * 8);
    }
  };

  v8f acc[3][4] = {};
  // A fragment: lane = row m0+l15; element e -> K = (e<8?e:e+8) + hi*8
  const _Float16* ap = xh + (size_t)(m0 + l15) * HID + hi * 8;

  fill(0, 0);
  for (int kb = 0; kb < HID; kb += 32) {
    const int buf = (kb >> 5) & 1;
    if (kb + 32 < HID) { fill(buf ^ 1, kb + 32); WAIT_ASYNC(3); }
    else               { WAIT_ASYNC(0); }
    __syncthreads();                       // slice `buf` visible to all

    Frag16 a;
    a.u[0] = *(const uint4*)(ap + kb);
    a.u[1] = *(const uint4*)(ap + kb + 16);
    if (kb + 32 < HID) __builtin_prefetch(ap + kb + 32, 0, 3);
#pragma unroll
    for (int p = 0; p < 3; ++p) {
#pragma unroll
      for (int t = 0; t < 4; ++t) {
        // B fragment from LDS: lane = col, element e -> K = (e<8?e:e+8) + hi*8
        Frag16 b;
        const _Float16* bp = &wl[buf][p][t * 16 + l15][hi * 8];
        b.u[0] = *(const uint4*)(bp);
        b.u[1] = *(const uint4*)(bp + 16);
        acc[p][t] = wmma_f16(a, b, acc[p][t]);
      }
    }
    __syncthreads();                       // done reading before refill
  }

  // Epilogue: bias + RoPE (+0.125 scale on Q). C layout: VGPR r -> row
  // m0 + r + 8*hi, column = n0 + t*16 + l15. RoPE partner = lane^1.
  const int h = ng;
#pragma unroll
  for (int t = 0; t < 4; ++t) {
    const int d = t * 16 + l15;        // 0..63 within head
    const int n = h * 64 + d;
    const int j = d >> 1;
    const float sgn = (d & 1) ? 1.0f : -1.0f;
    const float biq = bq[n], bik = bk[n], biv = bv[n];
#pragma unroll
    for (int r = 0; r < 8; ++r) {
      const int row = m0 + r + 8 * hi;
      const int bb  = row >> 10;           // / SEQ
      const int s   = row & (SEQ - 1);
      const float c  = cosc[s * 32 + j];
      const float sn = sinc[s * 32 + j];

      float vq = acc[0][t][r] + biq;
      float pq = __shfl_xor(vq, 1);
      float oq = (vq * c + sgn * pq * sn) * 0.125f;   // 1/sqrt(64)
      qr[((size_t)(bb * NHEAD + h) * SEQ + s) * HD + d] = (_Float16)oq;

      float vk = acc[1][t][r] + bik;
      float pk = __shfl_xor(vk, 1);
      float ok = vk * c + sgn * pk * sn;
      kr[((size_t)(bb * NHEAD + h) * SEQ + s) * HD + d] = (_Float16)ok;

      float vv = acc[2][t][r] + biv;
      vt[((size_t)(bb * NHEAD + h) * HD + d) * SEQ + s] = (_Float16)vv;
    }
  }
}

// ------------------------------------------------- RSE attention scan kernel
// Block = 8 waves covering 8 query-tiles of the SAME (b,h): each 32-key
// K-block (4KB) and V-block (4KB) is async-staged to LDS once per block,
// double-buffered, shared by all 8 waves. Sequential scan over 32 blocks.
__global__ __launch_bounds__(256) void attn_kernel(
    const _Float16* __restrict__ qr, const _Float16* __restrict__ kr,
    const _Float16* __restrict__ vt, const float* __restrict__ lamp,
    _Float16* __restrict__ ctx) {
  __shared__ __align__(16) _Float16 kl[2][32][64];   // [key][d]   8 KB
  __shared__ __align__(16) _Float16 vl[2][64][32];   // [d][key]   8 KB
  __shared__ __align__(16) _Float16 wbuf[8][16][32]; // per-wave w 8 KB

  const int tid  = threadIdx.x;
  const int lane = tid & 31;
  const int wid  = tid >> 5;
  const int bh   = blockIdx.x >> 3;                 // b*NHEAD + h
  const int qt   = (blockIdx.x & 7) * 8 + wid;      // query tile in [0,64)
  const int l15  = lane & 15;
  const int hi   = lane >> 4;
  const int q0   = qt * 16;
  const float lam = lamp[0];

  const _Float16* qb = qr + (size_t)bh * SEQ * HD;
  const _Float16* kb = kr + (size_t)bh * SEQ * HD;
  const _Float16* vb = vt + (size_t)bh * HD * SEQ;

  auto fill = [&](int buf, int k0) {
    // K block: key = tid>>3, chunk = tid&7  (32 x 64 halves)
    async_copy16(&kl[buf][tid >> 3][(tid & 7) * 8],
                 kb + (size_t)(k0 + (tid >> 3)) * HD + (tid & 7) * 8);
    // V block: d = tid>>2, chunk = tid&3    (64 x 32 halves)
    async_copy16(&vl[buf][tid >> 2][(tid & 3) * 8],
                 vb + (size_t)(tid >> 2) * SEQ + k0 + (tid & 3) * 8);
  };

  // Q A-fragments for d-steps [0,32) and [32,64)
  Frag16 aq0, aq1;
  {
    const _Float16* qp = qb + (size_t)(q0 + l15) * HD + hi * 8;
    aq0.u[0] = *(const uint4*)(qp);
    aq0.u[1] = *(const uint4*)(qp + 16);
    aq1.u[0] = *(const uint4*)(qp + 32);
    aq1.u[1] = *(const uint4*)(qp + 48);
  }

  v8f acc[4] = {};
  float rem[8], li[8], mi[8];
#pragma unroll
  for (int r = 0; r < 8; ++r) {
    rem[r] = 1.0f; li[r] = 0.0f; mi[r] = -__builtin_huge_valf();
  }

  fill(0, 0);
  for (int j = 0; j < SEQ / 32; ++j) {
    const int k0  = j * 32;
    const int buf = j & 1;
    if (j + 1 < SEQ / 32) { fill(buf ^ 1, k0 + 32); WAIT_ASYNC(2); }
    else                  { WAIT_ASYNC(0); }
    __syncthreads();                     // K/V block `buf` ready

    // logits = q . k^T for keys [k0,k0+16) and [k0+16,k0+32)
    v8f c0 = {}, c1 = {};
#pragma unroll
    for (int t = 0; t < 2; ++t) {
      v8f cc = {};
      Frag16 b;
      const _Float16* kp = &kl[buf][t * 16 + l15][hi * 8];
      b.u[0] = *(const uint4*)(kp);      b.u[1] = *(const uint4*)(kp + 16);
      cc = wmma_f16(aq0, b, cc);
      b.u[0] = *(const uint4*)(kp + 32); b.u[1] = *(const uint4*)(kp + 48);
      cc = wmma_f16(aq1, b, cc);
      if (t == 0) c0 = cc; else c1 = cc;
    }

    // elementwise: decay, clip, sigmoid, w = beta*rem, row reductions
#pragma unroll
    for (int r = 0; r < 8; ++r) {
      const float mpos = (float)(q0 + r + 8 * hi);
      const float p0   = (float)(k0 + l15);
      const float lg0 = c0[r] - lam * fabsf(mpos - p0);
      const float lg1 = c1[r] - lam * fabsf(mpos - p0 - 16.0f);
      const float t0 = fminf(fmaxf(lg0, -20.0f), 20.0f);
      const float t1 = fminf(fmaxf(lg1, -20.0f), 20.0f);
      const float w0 = rem[r] / (1.0f + __expf(-t0));
      const float w1 = rem[r] / (1.0f + __expf(-t1));
      float cons = w0 + w1;
      float mx   = fmaxf(lg0, lg1);
#pragma unroll
      for (int msk = 1; msk < 16; msk <<= 1) {   // stays inside 16-lane halves
        cons += __shfl_xor(cons, msk);
        mx = fmaxf(mx, __shfl_xor(mx, msk));
      }
      const float mnew = fmaxf(mi[r], mx);
      li[r] = li[r] * exp2f(mi[r] - mnew) + cons;
      mi[r] = mnew;
      rem[r] = fmaxf(rem[r] * (1.0f - cons), 1e-6f);
      wbuf[wid][r + 8 * hi][l15]      = (_Float16)w0;
      wbuf[wid][r + 8 * hi][l15 + 16] = (_Float16)w1;
    }
    __syncthreads();                     // wbuf visible (cross-lane via LDS)

    // reshape w (C layout) -> A fragment 16x32, then acc += w x V
    Frag16 aw;
    {
      const _Float16* wp = &wbuf[wid][l15][hi * 8];
      aw.u[0] = *(const uint4*)(wp);
      aw.u[1] = *(const uint4*)(wp + 16);
    }
#pragma unroll
    for (int t = 0; t < 4; ++t) {
      Frag16 b;
      const _Float16* vp = &vl[buf][t * 16 + l15][hi * 8];
      b.u[0] = *(const uint4*)(vp);
      b.u[1] = *(const uint4*)(vp + 16);
      acc[t] = wmma_f16(aw, b, acc[t]);
    }
    __syncthreads();                     // done reading before refill
  }

  const int bbat = bh / NHEAD, h = bh % NHEAD;
#pragma unroll
  for (int t = 0; t < 4; ++t)
#pragma unroll
    for (int r = 0; r < 8; ++r) {
      const int s = q0 + r + 8 * hi;
      const float o = acc[t][r] / fmaxf(li[r], 1e-6f);
      ctx[((size_t)(bbat * SEQ + s)) * HID + h * 64 + t * 16 + l15] =
          (_Float16)o;
    }
}

// --------------------------------------------------- output projection GEMM
__global__ __launch_bounds__(256) void out_proj_kernel(
    const _Float16* __restrict__ ctx, const _Float16* __restrict__ wo,
    const float* __restrict__ bo, float* __restrict__ out) {
  __shared__ __align__(16) _Float16 wl[2][64][32];   // 8 KB

  const int tid  = threadIdx.x;
  const int lane = tid & 31;
  const int wid  = tid >> 5;
  const int ng   = blockIdx.x % (HID / 64);
  const int m0   = (blockIdx.x / (HID / 64) * 8 + wid) * 16;
  const int n0   = ng * 64;
  const int l15  = lane & 15;
  const int hi   = lane >> 4;

  auto fill = [&](int buf, int kb) {
    const int nl = tid >> 2, ch = tid & 3;
    async_copy16(&wl[buf][nl][ch * 8],
                 wo + (size_t)(n0 + nl) * HID + kb + ch * 8);
  };

  v8f acc[4] = {};
  const _Float16* ap = ctx + (size_t)(m0 + l15) * HID + hi * 8;

  fill(0, 0);
  for (int kb = 0; kb < HID; kb += 32) {
    const int buf = (kb >> 5) & 1;
    if (kb + 32 < HID) { fill(buf ^ 1, kb + 32); WAIT_ASYNC(1); }
    else               { WAIT_ASYNC(0); }
    __syncthreads();

    Frag16 a;
    a.u[0] = *(const uint4*)(ap + kb);
    a.u[1] = *(const uint4*)(ap + kb + 16);
    if (kb + 32 < HID) __builtin_prefetch(ap + kb + 32, 0, 3);
#pragma unroll
    for (int t = 0; t < 4; ++t) {
      Frag16 b;
      const _Float16* bp = &wl[buf][t * 16 + l15][hi * 8];
      b.u[0] = *(const uint4*)(bp);
      b.u[1] = *(const uint4*)(bp + 16);
      acc[t] = wmma_f16(a, b, acc[t]);
    }
    __syncthreads();
  }
#pragma unroll
  for (int t = 0; t < 4; ++t) {
    const int n = n0 + t * 16 + l15;
    const float bias = bo[n];
#pragma unroll
    for (int r = 0; r < 8; ++r)
      out[(size_t)(m0 + r + 8 * hi) * HID + n] = acc[t][r] + bias;
  }
}

// ---------------------------------------------------------------------- host
extern "C" void kernel_launch(void* const* d_in, const int* in_sizes, int n_in,
                              void* d_out, int out_size, void* d_ws,
                              size_t ws_size, hipStream_t stream) {
  const float* x    = (const float*)d_in[0];
  const float* Wq   = (const float*)d_in[1];
  const float* bq   = (const float*)d_in[2];
  const float* Wk   = (const float*)d_in[3];
  const float* bk   = (const float*)d_in[4];
  const float* Wv   = (const float*)d_in[5];
  const float* bv   = (const float*)d_in[6];
  const float* Wo   = (const float*)d_in[7];
  const float* bo   = (const float*)d_in[8];
  const float* lam  = (const float*)d_in[9];
  const float* cosc = (const float*)d_in[10];
  const float* sinc = (const float*)d_in[11];

  const size_t NX = (size_t)MTOT * HID;   // 12,582,912 elems
  const size_t NW = (size_t)HID * HID;    // 589,824 elems
  _Float16* xh  = (_Float16*)d_ws;        // workspace: ~131 MB total (f16)
  _Float16* wqh = xh + NX;
  _Float16* wkh = wqh + NW;
  _Float16* wvh = wkh + NW;
  _Float16* woh = wvh + NW;
  _Float16* qr  = woh + NW;
  _Float16* kr  = qr + NX;
  _Float16* vt  = kr + NX;
  _Float16* ctx = vt + NX;

  cvt_f32_f16<<<(int)(NX / 4 / 256), 256, 0, stream>>>(x, xh, (int)(NX / 4));
  cvt_f32_f16<<<(int)(NW / 4 / 256), 256, 0, stream>>>(Wq, wqh, (int)(NW / 4));
  cvt_f32_f16<<<(int)(NW / 4 / 256), 256, 0, stream>>>(Wk, wkh, (int)(NW / 4));
  cvt_f32_f16<<<(int)(NW / 4 / 256), 256, 0, stream>>>(Wv, wvh, (int)(NW / 4));
  cvt_f32_f16<<<(int)(NW / 4 / 256), 256, 0, stream>>>(Wo, woh, (int)(NW / 4));

  // 1024/8 M-blocks * 12 N-groups = 1536 blocks (8 waves each)
  qkv_rope_kernel<<<1536, 256, 0, stream>>>(xh, wqh, wkh, wvh, bq, bk, bv,
                                            cosc, sinc, qr, kr, vt);
  // 192 (b,h) * 8 qt-blocks = 1536 blocks
  attn_kernel<<<1536, 256, 0, stream>>>(qr, kr, vt, lam, ctx);
  out_proj_kernel<<<1536, 256, 0, stream>>>(ctx, woh, bo, (float*)d_out);
}
